// Conformer_16999480558323
// MI455X (gfx1250) — matrix operand order repaired and checked by reading, compiled, not run on hardware
//
#include <hip/hip_runtime.h>
#include <hip/hip_bf16.h>

// ---------------------------------------------------------------------------
// Conformer encoder forward for MI455X (gfx1250).
// All GEMMs run through v_wmma_f32_16x16x32_f16 (f16 inputs, f32 accumulate).
// ---------------------------------------------------------------------------

typedef __attribute__((ext_vector_type(16))) _Float16 v16h;
typedef __attribute__((ext_vector_type(8)))  float    v8f;

#define CB   8
#define CT   1024
#define CD   256
#define CDFF 2048
#define CH   4
#define CDK  64
#define CNPOS 2047
#define CKER 31

union FragU { v16h v; uint4 q[2]; };

__device__ inline v16h load_a_frag(const _Float16* Ah, int mi, int lane) {
  // A 16x32 f16 fragment (ISA 7.12.2): lanes 0-15 of row m hold K{0..7,16..23},
  // lanes 16-31 hold K{8..15,24..31}.
  int r    = mi * 16 + (lane & 15);
  int half = lane >> 4;
  const _Float16* p = Ah + r * 32 + half * 8;
  FragU f;
  f.q[0] = *reinterpret_cast<const uint4*>(p);
  f.q[1] = *reinterpret_cast<const uint4*>(p + 16);
  return f.v;
}

__device__ inline v16h load_b_frag(const _Float16* Bh, int ni, int lane) {
  // B 32x16 f16 fragment: lane holds one column, lanes 0-15 K{0..7,16..23}, lanes 16-31 K{8..15,24..31}.
  // Bh is stored [n][k] so a column is contiguous.
  int n = ni * 16 + (lane & 15);
  const _Float16* p = Bh + n * 32 + (lane >> 4) * 8;
  FragU f;
  f.q[0] = *reinterpret_cast<const uint4*>(p);
  f.q[1] = *reinterpret_cast<const uint4*>(p + 16);
  return f.v;
}

__device__ inline void store4h(_Float16* p, float4 v) {
  union { _Float16 h[4]; uint2 u; } t;
  t.h[0] = (_Float16)v.x; t.h[1] = (_Float16)v.y;
  t.h[2] = (_Float16)v.z; t.h[3] = (_Float16)v.w;
  *reinterpret_cast<uint2*>(p) = t.u;
}

// C[M,N] = epilogue( alpha * (A[M,K] . op(W) + bias) )  (+ C if accum)
// A row-major (stride lda).  W: !KMAJOR -> [N,K] (X @ W^T); KMAJOR -> [K,N].
// blockIdx.z batches with element strides sA/sW/sC (head batching).
template <int TN, bool KMAJOR>
__global__ __launch_bounds__(256)
void gemm_kernel(const float* __restrict__ A, int lda, size_t sA,
                 const float* __restrict__ W, int ldw, size_t sW,
                 const float* __restrict__ bias,
                 float* __restrict__ C, int ldc, size_t sC,
                 int M, int N, int K, float alpha, int accum, int relu)
{
  constexpr int NTW = TN / 32;              // 16-wide n-tiles per wave
  __shared__ __align__(16) _Float16 Ah[64 * 32];
  __shared__ __align__(16) _Float16 Bh[TN * 32];

  const float* Az = A + sA * blockIdx.z;
  const float* Wz = W + sW * blockIdx.z;
  float*       Cz = C + sC * blockIdx.z;

  int tid  = threadIdx.x;
  int lane = tid & 31;
  int w    = tid >> 5;                      // 8 waves
  int m0   = blockIdx.y * 64;
  int n0   = blockIdx.x * TN;
  int mi   = w & 3;                         // wave's 16-row slice
  int nj   = w >> 2;                        // wave's group of NTW col slices

  bool interior = (m0 + 64 <= M) && (n0 + TN <= N);

  v8f acc[NTW] = {};

  if (interior) {
    // ================= fast path: no guards inside the hot loop ===========
    for (int k0 = 0; k0 < K; k0 += 32) {
      #pragma unroll
      for (int e = 0; e < 2; ++e) {         // A: 64x32 floats as float4
        int lin = tid + e * 256;
        int ml  = lin >> 3;
        int kq  = (lin & 7) * 4;
        const float* gp = Az + (size_t)(m0 + ml) * lda + (k0 + kq);
        float4 vv = *reinterpret_cast<const float4*>(gp);
        if (k0 + 32 < K) __builtin_prefetch(gp + 32, 0, 0);  // global_prefetch
        store4h(&Ah[ml * 32 + kq], vv);
      }
      if constexpr (!KMAJOR) {
        #pragma unroll
        for (int e = 0; e < TN / 32; ++e) { // B: TNx32 floats as float4
          int lin = tid + e * 256;
          int nl  = lin >> 3;
          int kq  = (lin & 7) * 4;
          const float* gp = Wz + (size_t)(n0 + nl) * ldw + (k0 + kq);
          float4 vv = *reinterpret_cast<const float4*>(gp);
          if (k0 + 32 < K) __builtin_prefetch(gp + 32, 0, 0);
          store4h(&Bh[nl * 32 + kq], vv);
        }
      } else {
        #pragma unroll
        for (int e = 0; e < TN / 32; ++e) { // B: [K,N], vector along n, transpose
          int lin = tid + e * 256;
          int kk  = lin / (TN / 4);
          int nq  = (lin % (TN / 4)) * 4;
          const float* gp = Wz + (size_t)(k0 + kk) * ldw + (n0 + nq);
          float4 vv = *reinterpret_cast<const float4*>(gp);
          if (k0 + 32 < K) __builtin_prefetch(gp + (size_t)32 * ldw, 0, 0);
          Bh[(nq + 0) * 32 + kk] = (_Float16)vv.x;
          Bh[(nq + 1) * 32 + kk] = (_Float16)vv.y;
          Bh[(nq + 2) * 32 + kk] = (_Float16)vv.z;
          Bh[(nq + 3) * 32 + kk] = (_Float16)vv.w;
        }
      }
      __syncthreads();

      // Load ALL fragments first (distinct registers), then issue WMMAs
      // back-to-back so the ds loads pipeline and the matrix ops co-execute.
      v16h af = load_a_frag(Ah, mi, lane);
      v16h bf[NTW];
      #pragma unroll
      for (int j = 0; j < NTW; ++j) bf[j] = load_b_frag(Bh, nj * NTW + j, lane);
      #pragma unroll
      for (int j = 0; j < NTW; ++j)
        acc[j] = __builtin_amdgcn_wmma_f32_16x16x32_f16(false, af, false, bf[j],
                                                        (short)0, acc[j], false, false);
      __syncthreads();
    }
  } else {
    // ================= guarded path (only ragged 2047-edge tiles) =========
    for (int k0 = 0; k0 < K; k0 += 32) {
      #pragma unroll
      for (int e = 0; e < 8; ++e) {
        int lin = tid + e * 256;
        int ml  = lin >> 5;
        int kk  = lin & 31;
        int m   = m0 + ml;
        float vv = 0.0f;
        if (m < M) vv = Az[(size_t)m * lda + (k0 + kk)];
        Ah[ml * 32 + kk] = (_Float16)vv;
      }
      #pragma unroll
      for (int e = 0; e < TN / 8; ++e) {
        int lin = tid + e * 256;
        int nl  = lin >> 5;
        int kk  = lin & 31;
        int n   = n0 + nl;
        float vv = 0.0f;
        if (n < N) {
          if constexpr (!KMAJOR) vv = Wz[(size_t)n * ldw + (k0 + kk)];
          else                   vv = Wz[(size_t)(k0 + kk) * ldw + n];
        }
        Bh[nl * 32 + kk] = (_Float16)vv;
      }
      __syncthreads();

      v16h af = load_a_frag(Ah, mi, lane);
      v16h bf[NTW];
      #pragma unroll
      for (int j = 0; j < NTW; ++j) bf[j] = load_b_frag(Bh, nj * NTW + j, lane);
      #pragma unroll
      for (int j = 0; j < NTW; ++j)
        acc[j] = __builtin_amdgcn_wmma_f32_16x16x32_f16(false, af, false, bf[j],
                                                        (short)0, acc[j], false, false);
      __syncthreads();
    }
  }

  // ---- epilogue: D lane l holds col (l&15), rows (l>>4)*8 + r ----
  int mb = m0 + mi * 16 + (lane >> 4) * 8;
  #pragma unroll
  for (int j = 0; j < NTW; ++j) {
    int ncol = n0 + (nj * NTW + j) * 16 + (lane & 15);
    if (ncol >= N) continue;
    float bv = bias ? bias[ncol] : 0.0f;
    #pragma unroll
    for (int r = 0; r < 8; ++r) {
      int m = mb + r;
      if (m >= M) continue;
      float vv = (acc[j][r] + bv) * alpha;
      if (relu) vv = fmaxf(vv, 0.0f);
      size_t idx = (size_t)m * ldc + ncol;
      if (accum) vv += Cz[idx];
      Cz[idx] = vv;
    }
  }
}

// ---------------------------------------------------------------------------
// LayerNorm: one wave32 per row of 256 channels.
// ---------------------------------------------------------------------------
__global__ __launch_bounds__(256)
void ln_kernel(const float* __restrict__ x, float* __restrict__ y,
               const float* __restrict__ g, const float* __restrict__ b,
               int rows)
{
  int row  = blockIdx.x * 8 + (threadIdx.x >> 5);
  int lane = threadIdx.x & 31;
  if (row >= rows) return;
  const float* px = x + (size_t)row * CD;

  float v[8];
  float s = 0.0f;
  #pragma unroll
  for (int j = 0; j < 8; ++j) { v[j] = px[lane + 32 * j]; s += v[j]; }
  #pragma unroll
  for (int o = 16; o > 0; o >>= 1) s += __shfl_xor(s, o);
  float mu = s * (1.0f / CD);

  float s2 = 0.0f;
  #pragma unroll
  for (int j = 0; j < 8; ++j) { float d = v[j] - mu; s2 += d * d; }
  #pragma unroll
  for (int o = 16; o > 0; o >>= 1) s2 += __shfl_xor(s2, o);
  float rs = rsqrtf(s2 * (1.0f / CD) + 1e-12f);

  float* py = y + (size_t)row * CD;
  #pragma unroll
  for (int j = 0; j < 8; ++j) {
    int c = lane + 32 * j;
    py[c] = (v[j] - mu) * rs * g[c] + b[c];
  }
}

// ---------------------------------------------------------------------------
// Relative position table: pe[i,2j]=sin(r*div_j), pe[i,2j+1]=cos(r*div_j),
// r = T-1-i, i in [0, 2T-2].
// ---------------------------------------------------------------------------
__global__ __launch_bounds__(256)
void pos_kernel(float* __restrict__ pos)
{
  int idx = blockIdx.x * 256 + threadIdx.x;
  if (idx >= CNPOS * CD) return;
  int i = idx >> 8;
  int c = idx & 255;
  float r   = (float)(CT - 1 - i);
  float dv  = expf(-(float)(c & ~1) * (logf(10000.0f) / (float)CD));
  float ang = r * dv;
  pos[idx] = (c & 1) ? cosf(ang) : sinf(ang);
}

// ---------------------------------------------------------------------------
// scores[t,s] = (ac[t,s] + bd[t, T-1-t+s]) / 8, masked softmax over s.
// grid = (T, H); blockIdx.y selects the head slice of sc/bd.
// ---------------------------------------------------------------------------
__global__ __launch_bounds__(256)
void attn_softmax_kernel(float* __restrict__ sc_all, const float* __restrict__ bd_all,
                         const int* __restrict__ x_len, int b)
{
  int t   = blockIdx.x;
  int tid = threadIdx.x;
  int len = x_len[b];
  float* sc = sc_all + (size_t)blockIdx.y * CT * CT;
  const float* bd = bd_all + (size_t)blockIdx.y * CT * CNPOS;
  __shared__ float red[256];

  float vals[4];
  float mx = -3.4028235e38f;
  #pragma unroll
  for (int j = 0; j < 4; ++j) {
    int s = tid + j * 256;
    float vv = (sc[(size_t)t * CT + s] +
                bd[(size_t)t * CNPOS + (CT - 1 - t + s)]) * 0.125f;
    if (s >= len) vv = -3.4028235e38f;
    vals[j] = vv;
    mx = fmaxf(mx, vv);
  }
  red[tid] = mx; __syncthreads();
  for (int st = 128; st > 0; st >>= 1) {
    if (tid < st) red[tid] = fmaxf(red[tid], red[tid + st]);
    __syncthreads();
  }
  mx = red[0]; __syncthreads();

  float sum = 0.0f;
  #pragma unroll
  for (int j = 0; j < 4; ++j) {
    int s = tid + j * 256;
    vals[j] = (s < len) ? expf(vals[j] - mx) : 0.0f;
    sum += vals[j];
  }
  red[tid] = sum; __syncthreads();
  for (int st = 128; st > 0; st >>= 1) {
    if (tid < st) red[tid] += red[tid + st];
    __syncthreads();
  }
  float inv = 1.0f / red[0];
  #pragma unroll
  for (int j = 0; j < 4; ++j)
    sc[(size_t)t * CT + tid + j * 256] = vals[j] * inv;
}

// ---------------------------------------------------------------------------
// Elementwise helpers.
// ---------------------------------------------------------------------------
__global__ __launch_bounds__(256)
void addbias_kernel(const float* __restrict__ in, const float* __restrict__ bias,
                    float* __restrict__ out)
{
  size_t idx = (size_t)blockIdx.x * 256 + threadIdx.x;   // B*T*256
  out[idx] = in[idx] + bias[idx & 255];
}

__global__ __launch_bounds__(256)
void cat_kernel(const float* __restrict__ y, const float* __restrict__ a,
                float* __restrict__ cat)
{
  size_t idx = (size_t)blockIdx.x * 256 + threadIdx.x;   // B*T*512
  size_t row = idx >> 9;
  int    c   = (int)(idx & 511);
  cat[idx] = (c < CD) ? y[row * CD + c] : a[row * CD + (c - CD)];
}

__global__ __launch_bounds__(256)
void glu_kernel(const float* __restrict__ in, float* __restrict__ out)
{
  size_t idx = (size_t)blockIdx.x * 256 + threadIdx.x;   // B*T*256
  size_t row = idx >> 8;
  int    c   = (int)(idx & 255);
  float av = in[row * 512 + c];
  float gv = in[row * 512 + CD + c];
  out[idx] = av * (1.0f / (1.0f + expf(-gv)));
}

__global__ __launch_bounds__(256)
void dwconv_bn_kernel(const float* __restrict__ in, const float* __restrict__ w,
                      const float* __restrict__ dwb,
                      const float* __restrict__ bng, const float* __restrict__ bnb,
                      const float* __restrict__ bnm, const float* __restrict__ bnv,
                      float* __restrict__ out)
{
  size_t idx = (size_t)blockIdx.x * 256 + threadIdx.x;   // B*T*256
  int c = (int)(idx & 255);
  int t = (int)((idx >> 8) & (CT - 1));
  size_t bt0 = (idx >> 8) - t;                           // b*T
  float s = 0.0f;
  #pragma unroll
  for (int kk = 0; kk < CKER; ++kk) {
    int tt = t + kk - (CKER - 1) / 2;
    if (tt >= 0 && tt < CT)
      s += in[(bt0 + tt) * CD + c] * w[c * CKER + kk];
  }
  s += dwb[c];
  s = (s - bnm[c]) * rsqrtf(bnv[c] + 1e-5f) * bng[c] + bnb[c];
  out[idx] = fmaxf(s, 0.0f);
}

__global__ __launch_bounds__(256)
void out_kernel(const float* __restrict__ h, const int* __restrict__ x_len,
                float* __restrict__ out)
{
  size_t idx = (size_t)blockIdx.x * 256 + threadIdx.x;   // B*T*256
  int t = (int)((idx >> 8) & (CT - 1));
  int b = (int)(idx >> 18);                              // /(T*D)
  out[idx] = h[idx] * ((t < x_len[b]) ? 1.0f : 0.0f);
}

__global__ __launch_bounds__(256)
void mask_kernel(const int* __restrict__ x_len, float* __restrict__ m)
{
  int idx = blockIdx.x * 256 + threadIdx.x;              // B*T
  if (idx >= CB * CT) return;
  int b = idx / CT, t = idx % CT;
  m[idx] = (t < x_len[b]) ? 1.0f : 0.0f;
}

// ---------------------------------------------------------------------------
// Host orchestration.
// ---------------------------------------------------------------------------
enum {
  P_BIAS_U = 0, P_BIAS_V, P_BK, P_BN_B, P_BN_G, P_BN_M, P_BN_V, P_BO, P_BQ, P_BV,
  P_CONCAT_B, P_CONCAT_W, P_DW_B, P_DW_W, P_FF_B1, P_FF_B2, P_FF_W1, P_FF_W2,
  P_FFM_B1, P_FFM_B2, P_FFM_W1, P_FFM_W2, P_LN_CONV_B, P_LN_CONV_G, P_LN_FF_B,
  P_LN_FF_G, P_LN_FFM_B, P_LN_FFM_G, P_LN_FINAL_B, P_LN_FINAL_G, P_LN_MHA_B,
  P_LN_MHA_G, P_PW1_B, P_PW1_W, P_PW2_B, P_PW2_W, P_WK, P_WO, P_WPOS, P_WQ, P_WV,
  P_COUNT
};

static const size_t PSTRIDE[P_COUNT] = {
  256, 256, 256, 256, 256, 256, 256, 256, 256, 256,
  256, (size_t)256 * 512, 256, (size_t)256 * 31, 2048, 256,
  (size_t)2048 * 256, (size_t)256 * 2048, 2048, 256,
  (size_t)2048 * 256, (size_t)256 * 2048,
  256, 256, 256, 256, 256, 256, 256, 256, 256, 256,
  512, (size_t)512 * 256, 256, (size_t)256 * 256,
  (size_t)256 * 256, (size_t)256 * 256, (size_t)256 * 256,
  (size_t)256 * 256, (size_t)256 * 256
};

static inline void gemm(hipStream_t s,
                        const float* A, int lda, size_t sA,
                        const float* W, int ldw, size_t sW, int kmajor,
                        const float* bias,
                        float* C, int ldc, size_t sC,
                        int M, int N, int K, float alpha, int accum, int relu,
                        int batches = 1)
{
  if (kmajor) {
    dim3 g((N + 63) / 64, (M + 63) / 64, batches);
    gemm_kernel<64, true><<<g, 256, 0, s>>>(A, lda, sA, W, ldw, sW, bias,
                                            C, ldc, sC, M, N, K, alpha, accum, relu);
  } else {
    dim3 g((N + 127) / 128, (M + 63) / 64, batches);
    gemm_kernel<128, false><<<g, 256, 0, s>>>(A, lda, sA, W, ldw, sW, bias,
                                              C, ldc, sC, M, N, K, alpha, accum, relu);
  }
}

extern "C" void kernel_launch(void* const* d_in, const int* in_sizes, int n_in,
                              void* d_out, int out_size, void* d_ws, size_t ws_size,
                              hipStream_t stream)
{
  (void)in_sizes; (void)n_in; (void)out_size; (void)ws_size;

  const float* x       = (const float*)d_in[0];
  const int*   x_len   = (const int*)d_in[1];
  const float* embed_b = (const float*)d_in[2];
  const float* embed_w = (const float*)d_in[3];
  const float* const* LP = (const float* const*)(d_in + 4);

  const size_t BT   = (size_t)CB * CT;        // 8192
  const size_t nBTD = BT * CD;                // 2,097,152

  float* ws  = (float*)d_ws;
  float* h   = ws;                            // [B,T,D]
  float* y   = h   + nBTD;                    // [B,T,D]
  float* big = y   + nBTD;                    // [B,T,DFF] multi-use
  float* q   = big + BT * CDFF;               // [B,T,D] (also GLU out)
  float* k   = q   + nBTD;                    // [B,T,D] (also conv out)
  float* v   = k   + nBTD;                    // [B,T,D]
  float* a   = v   + nBTD;                    // [B,T,D] attention output
  float* qu  = a   + nBTD;                    // [B,T,D] q + bias_u
  float* qv  = qu  + nBTD;                    // [B,T,D] q + bias_v
  float* pm  = qv  + nBTD;                    // [2047,256]
  float* pos = pm  + (size_t)2048 * 256;      // [2047,256]
  float* sc  = big;                           // [H,T,T]    (attention phase)
  float* bd  = big + (size_t)CH * CT * CT;    // [H,T,2047] (attention phase)

  pos_kernel<<<(CNPOS * CD + 255) / 256, 256, 0, stream>>>(pos);
  gemm(stream, x, 1024, 0, embed_w, 1024, 0, 0, embed_b,
       h, CD, 0, (int)BT, CD, 1024, 16.0f, 0, 0);

  const int NBLK = (int)(nBTD / 256);

  for (int l = 0; l < 6; ++l) {
    const float* PW[P_COUNT];
    for (int i = 0; i < P_COUNT; ++i) PW[i] = LP[i] + PSTRIDE[i] * l;

    // ---- Macaron feed-forward: x += 0.5 * FF(LN(x)) ----
    ln_kernel<<<(int)(BT / 8), 256, 0, stream>>>(h, y, PW[P_LN_FFM_G], PW[P_LN_FFM_B], (int)BT);
    gemm(stream, y, CD, 0, PW[P_FFM_W1], CD, 0, 0, PW[P_FFM_B1],
         big, CDFF, 0, (int)BT, CDFF, CD, 1.0f, 0, 1);
    gemm(stream, big, CDFF, 0, PW[P_FFM_W2], CDFF, 0, 0, PW[P_FFM_B2],
         h, CD, 0, (int)BT, CD, CDFF, 0.5f, 1, 0);

    // ---- Relative-position multi-head attention ----
    ln_kernel<<<(int)(BT / 8), 256, 0, stream>>>(h, y, PW[P_LN_MHA_G], PW[P_LN_MHA_B], (int)BT);
    gemm(stream, y, CD, 0, PW[P_WQ], CD, 0, 0, PW[P_BQ], q, CD, 0, (int)BT, CD, CD, 1.0f, 0, 0);
    gemm(stream, y, CD, 0, PW[P_WK], CD, 0, 0, PW[P_BK], k, CD, 0, (int)BT, CD, CD, 1.0f, 0, 0);
    gemm(stream, y, CD, 0, PW[P_WV], CD, 0, 0, PW[P_BV], v, CD, 0, (int)BT, CD, CD, 1.0f, 0, 0);
    gemm(stream, pos, CD, 0, PW[P_WPOS], CD, 0, 0, nullptr,
         pm, CD, 0, CNPOS, CD, CD, 1.0f, 0, 0);
    addbias_kernel<<<NBLK, 256, 0, stream>>>(q, PW[P_BIAS_U], qu);  // bias_u flattens to [256]
    addbias_kernel<<<NBLK, 256, 0, stream>>>(q, PW[P_BIAS_V], qv);

    for (int b = 0; b < CB; ++b) {
      size_t bo = (size_t)b * CT * CD;
      // ac[h] = (q + bias_u) . k^T   (head-batched via grid.z, strides of 64)
      gemm(stream, qu + bo, CD, CDK, k + bo, CD, CDK, 0, nullptr,
           sc, CT, (size_t)CT * CT, CT, CT, CDK, 1.0f, 0, 0, CH);
      // bd[h] = (q + bias_v) . pm_h^T
      gemm(stream, qv + bo, CD, CDK, pm, CD, CDK, 0, nullptr,
           bd, CNPOS, (size_t)CT * CNPOS, CT, CNPOS, CDK, 1.0f, 0, 0, CH);
      // masked softmax of (ac + rel_shift(bd)) / sqrt(dk), all heads
      attn_softmax_kernel<<<dim3(CT, CH), 256, 0, stream>>>(sc, bd, x_len, b);
      // out[h] = attn . v  (B matrix K-major)
      gemm(stream, sc, CT, (size_t)CT * CT, v + bo, CD, CDK, 1, nullptr,
           a + bo, CD, CDK, CT, CDK, CT, 1.0f, 0, 0, CH);
    }
    // x += concat([y, a]) @ concat_w^T + concat_b
    cat_kernel<<<(int)(BT * 512 / 256), 256, 0, stream>>>(y, a, big);
    gemm(stream, big, 512, 0, PW[P_CONCAT_W], 512, 0, 0, PW[P_CONCAT_B],
         h, CD, 0, (int)BT, CD, 512, 1.0f, 1, 0);

    // ---- Convolution module ----
    ln_kernel<<<(int)(BT / 8), 256, 0, stream>>>(h, y, PW[P_LN_CONV_G], PW[P_LN_CONV_B], (int)BT);
    gemm(stream, y, CD, 0, PW[P_PW1_W], CD, 0, 0, PW[P_PW1_B],
         big, 512, 0, (int)BT, 512, CD, 1.0f, 0, 0);
    glu_kernel<<<NBLK, 256, 0, stream>>>(big, q);                 // q := GLU out
    dwconv_bn_kernel<<<NBLK, 256, 0, stream>>>(q, PW[P_DW_W], PW[P_DW_B],
        PW[P_BN_G], PW[P_BN_B], PW[P_BN_M], PW[P_BN_V], k);       // k := conv out
    gemm(stream, k, CD, 0, PW[P_PW2_W], CD, 0, 0, PW[P_PW2_B],
         h, CD, 0, (int)BT, CD, CD, 1.0f, 1, 0);

    // ---- Second feed-forward: x += 0.5 * FF(LN(x)) ----
    ln_kernel<<<(int)(BT / 8), 256, 0, stream>>>(h, y, PW[P_LN_FF_G], PW[P_LN_FF_B], (int)BT);
    gemm(stream, y, CD, 0, PW[P_FF_W1], CD, 0, 0, PW[P_FF_B1],
         big, CDFF, 0, (int)BT, CDFF, CD, 1.0f, 0, 1);
    gemm(stream, big, CDFF, 0, PW[P_FF_W2], CDFF, 0, 0, PW[P_FF_B2],
         h, CD, 0, (int)BT, CD, CDFF, 0.5f, 1, 0);

    // ---- Final per-layer LayerNorm (in place; wave reads full row first) ----
    ln_kernel<<<(int)(BT / 8), 256, 0, stream>>>(h, h, PW[P_LN_FINAL_G], PW[P_LN_FINAL_B], (int)BT);
  }

  // ---- Output: h * mask, then mask itself (tuple flattened) ----
  out_kernel<<<NBLK, 256, 0, stream>>>(h, x_len, (float*)d_out);
  mask_kernel<<<(int)((BT + 255) / 256), 256, 0, stream>>>(x_len, (float*)d_out + nBTD);
}
